// Attention_22256520528709
// MI455X (gfx1250) — matrix operand order repaired and checked by reading, not compiled, not run on hardware
//
#include <hip/hip_runtime.h>
#include <hip/hip_bf16.h>
#include <stdint.h>

// ---------------------------------------------------------------------------
// Attention block for MI455X (gfx1250, wave32, WMMA + TDM).
// B=4, T=2048, DIM=1024, H=16, DH=64.  All GEMMs in bf16 WMMA, fp32 accum.
// ---------------------------------------------------------------------------

typedef __bf16 bf16;
typedef __bf16 v16bf __attribute__((ext_vector_type(16)));
typedef float  v8f   __attribute__((ext_vector_type(8)));
typedef unsigned int u32x4 __attribute__((ext_vector_type(4)));
typedef int          i32x4 __attribute__((ext_vector_type(4)));
typedef int          i32x8 __attribute__((ext_vector_type(8)));

#if defined(__has_builtin)
#  if __has_builtin(__builtin_amdgcn_tensor_load_to_lds) && \
      __has_builtin(__builtin_amdgcn_s_wait_tensorcnt)
#    define USE_TDM 1
#  endif
#endif
#ifndef USE_TDM
#  define USE_TDM 0
#endif

union FragU { v16bf v; uint4 q[2]; };

__device__ __forceinline__ v8f zero8() {
  v8f z = {0.f, 0.f, 0.f, 0.f, 0.f, 0.f, 0.f, 0.f};
  return z;
}

__device__ __forceinline__ v8f wmma_bf16(v16bf a, v16bf b, v8f c) {
  // D = A(16x32 bf16) * B(32x16 bf16) + C(16x16 f32)
  return __builtin_amdgcn_wmma_f32_16x16x32_bf16(
      /*neg_a=*/false, a, /*neg_b=*/false, b,
      /*c_mod=*/(short)0, c, /*reuse_a=*/false, /*reuse_b=*/false);
}

// A fragment: 16x32 bf16 tile, row-major in LDS with `stride` elements/row.
// Lanes 0-15: M=lane, K=[0..7]+[16..23]; lanes 16-31: M=lane-16, K=[8..15]+[24..31].
__device__ __forceinline__ v16bf load_frag_a(const bf16* base, int stride,
                                             int lane, int kk) {
  int m  = lane & 15;
  int bk = (lane >> 4) << 3;                 // 0 or 8
  const bf16* p = base + m * stride + kk + bk;
  FragU f;
  f.q[0] = *(const uint4*)(p);
  f.q[1] = *(const uint4*)(p + 16);
  return f.v;
}

// B fragment: 32x16 bf16, B[k][n] = Wrow[n][k] (row-major by n).
// Lanes 0-15: K=[0..7]+[16..23]; lanes 16-31: K=[8..15]+[24..31].
__device__ __forceinline__ v16bf load_frag_b(const bf16* base, int stride,
                                             int lane, int kk) {
  int n  = lane & 15;
  int bk = (lane >> 4) << 3;                 // 0 or 8
  const bf16* p = base + n * stride + kk + bk;
  FragU f;
  f.q[0] = *(const uint4*)(p);
  f.q[1] = *(const uint4*)(p + 16);
  return f.v;
}

#if USE_TDM
// TDM 2D tile load: global (row-major, `stride` elems of bf16 between rows)
// -> LDS with pad_amount = 4 DWORDs (8 bf16) appended per row.
// pad_interval code: row bytes = 2^(code+3)  (code 3 -> 32 bf16, code 4 -> 64 bf16)
__device__ __forceinline__ void tdm_load_2d(void* lds_dst, const void* gsrc,
                                            int tile_w, int tile_h,
                                            long long stride,
                                            int pad_interval) {
  unsigned int lds_addr = (unsigned int)(uintptr_t)lds_dst;
  unsigned long long ga = (unsigned long long)(uintptr_t)gsrc;
  u32x4 g0;
  g0[0] = 1u;                                   // count=1, user load
  g0[1] = lds_addr;                             // LDS byte address
  g0[2] = (unsigned int)ga;                     // global addr lo
  g0[3] = (unsigned int)((ga >> 32) & 0x01FFFFFFu) | (2u << 30);  // type=2
  i32x8 g1;
  g1[0] = (1 << 16)                             // data_size = 2 bytes
        | (1 << 20)                             // pad_enable
        | (pad_interval << 22)                  // DWORDs between pads
        | (3 << 25);                            // pad_amount = 4 DWORDs
  g1[1] = (tile_w & 0xFFFF) << 16;              // tensor_dim0[15:0]
  g1[2] = (tile_h & 0xFFFF) << 16;              // tensor_dim0 hi | tensor_dim1 lo
  g1[3] = (tile_w & 0xFFFF) << 16;              // tensor_dim1 hi | tile_dim0
  g1[4] = (tile_h & 0xFFFF);                    // tile_dim1 | tile_dim2=0
  g1[5] = (int)(stride & 0xFFFFFFFFll);         // tensor_dim0_stride lo
  g1[6] = (int)((stride >> 32) & 0xFFFFll);     // stride hi | dim1_stride=0
  g1[7] = 0;
  i32x4 g2 = {0, 0, 0, 0};
  i32x4 g3 = {0, 0, 0, 0};
  i32x8 g4 = {0, 0, 0, 0, 0, 0, 0, 0};          // aux (6-arg toolchain form)
  __builtin_amdgcn_tensor_load_to_lds(g0, g1, g2, g3, g4, 0);
}
#endif

// ---------------------------------------------------------------------------
// fp32 -> bf16 conversion
// ---------------------------------------------------------------------------
__global__ void cvt_bf16_kernel(const float* __restrict__ src,
                                bf16* __restrict__ dst, int n) {
  int i = blockIdx.x * blockDim.x + threadIdx.x;
  if (i < n) dst[i] = (bf16)src[i];
}

// ---------------------------------------------------------------------------
// Shared GEMM mainloop:  C[128x128] = A[M,K] * W[N,K]^T   (block tile)
// 8 waves: 4 in M (32 rows) x 2 in N (64 cols); BK = 32; LDS stride 40.
// ---------------------------------------------------------------------------
__device__ __forceinline__ void gemm_mainloop(const bf16* __restrict__ A,
                                              const bf16* __restrict__ W,
                                              int K, int m0, int n0,
                                              bf16* as, bf16* bs,
                                              v8f acc[2][4]) {
  const int tid  = threadIdx.x;
  const int lane = tid & 31;
  const int wave = tid >> 5;
  const int wm = wave >> 1;                  // 0..3
  const int wn = wave & 1;                   // 0..1

#pragma unroll
  for (int mi = 0; mi < 2; ++mi)
#pragma unroll
    for (int nj = 0; nj < 4; ++nj) acc[mi][nj] = zero8();

  for (int k0 = 0; k0 < K; k0 += 32) {
    __syncthreads();
#if USE_TDM
    if (wave == 0) {
      tdm_load_2d(as, A + (size_t)m0 * K + k0, 32, 128, K, 3);
      tdm_load_2d(bs, W + (size_t)n0 * K + k0, 32, 128, K, 3);
      __builtin_amdgcn_s_wait_tensorcnt(0);
    }
#else
    // stage A tile (128x32) and W tile (128x32): 512 16B chunks each
#pragma unroll
    for (int it = 0; it < 2; ++it) {
      int i = tid + it * 256;
      int r = i >> 2, c = i & 3;
      *(uint4*)(as + r * 40 + c * 8) =
          *(const uint4*)(A + (size_t)(m0 + r) * K + k0 + c * 8);
      *(uint4*)(bs + r * 40 + c * 8) =
          *(const uint4*)(W + (size_t)(n0 + r) * K + k0 + c * 8);
      if (k0 + 32 < K) {
        __builtin_prefetch(A + (size_t)(m0 + r) * K + k0 + 32 + c * 8, 0, 1);
        __builtin_prefetch(W + (size_t)(n0 + r) * K + k0 + 32 + c * 8, 0, 1);
      }
    }
#endif
    __syncthreads();

    v16bf afr[2], bfr[4];
#pragma unroll
    for (int mi = 0; mi < 2; ++mi)
      afr[mi] = load_frag_a(as + (wm * 32 + mi * 16) * 40, 40, lane, 0);
#pragma unroll
    for (int nj = 0; nj < 4; ++nj)
      bfr[nj] = load_frag_b(bs + (wn * 64 + nj * 16) * 40, 40, lane, 0);
#pragma unroll
    for (int mi = 0; mi < 2; ++mi)
#pragma unroll
      for (int nj = 0; nj < 4; ++nj)
        acc[mi][nj] = wmma_bf16(afr[mi], bfr[nj], acc[mi][nj]);
  }
}

// ---------------------------------------------------------------------------
// QKV projection:  qkv[m, e] = X[m,:] . Wqkv[e,:],  e = d*48 + k*16 + h
// Scatter directly into q/k/v bf16 buffers laid out [B,H,T,DH].
// ---------------------------------------------------------------------------
__global__ __launch_bounds__(256) void qkv_gemm_kernel(
    const bf16* __restrict__ X, const bf16* __restrict__ Wq,
    bf16* __restrict__ qb, bf16* __restrict__ kb, bf16* __restrict__ vb) {
  __shared__ __align__(16) bf16 as[128 * 40];
  __shared__ __align__(16) bf16 bs[128 * 40];
  v8f acc[2][4];
  const int m0 = blockIdx.x * 128;
  const int n0 = blockIdx.y * 128;
  gemm_mainloop(X, Wq, 1024, m0, n0, as, bs, acc);

  const int lane = threadIdx.x & 31;
  const int wave = threadIdx.x >> 5;
  const int wm = wave >> 1, wn = wave & 1;
  const int h = lane & 15;                   // e % 16 within 16-aligned tile
  const int mhi = (lane >> 4) << 3;          // +8 rows for lanes 16-31
#pragma unroll
  for (int mi = 0; mi < 2; ++mi) {
#pragma unroll
    for (int nj = 0; nj < 4; ++nj) {
      int e_tile = n0 + wn * 64 + nj * 16;   // multiple of 16
      int d   = e_tile / 48;
      int kk3 = (e_tile % 48) >> 4;          // 0=q, 1=k, 2=v
      bf16* dst = (kk3 == 0) ? qb : (kk3 == 1) ? kb : vb;
#pragma unroll
      for (int r = 0; r < 8; ++r) {
        int m = m0 + wm * 32 + mi * 16 + r + mhi;
        int b = m >> 11, t = m & 2047;
        dst[(size_t)((b * 16 + h) * 2048 + t) * 64 + d] =
            (bf16)acc[mi][nj][r];
      }
    }
  }
}

// ---------------------------------------------------------------------------
// Flash attention: one block = 128 query rows of one (b,h). 8 waves x 16 rows.
// Key blocks of 64, online softmax, O accumulated in fp32 via WMMA.
// ---------------------------------------------------------------------------
__global__ __launch_bounds__(256) void attn_kernel(
    const bf16* __restrict__ qg, const bf16* __restrict__ kg,
    const bf16* __restrict__ vg, bf16* __restrict__ og) {
  __shared__ __align__(16) bf16 qs[128 * 72];    // Q tile  [row][d]
  __shared__ __align__(16) bf16 ks[64 * 72];     // K tile  [j][d]
  __shared__ __align__(16) bf16 vt[64 * 72];     // V^T     [d][j]
  __shared__ __align__(16) bf16 ps[8 * 16 * 72]; // per-wave P tiles

  const int tid = threadIdx.x, lane = tid & 31, wave = tid >> 5;
  const int bh = blockIdx.y;
  const int q0 = blockIdx.x * 128;
  const size_t base = (size_t)bh * 2048 * 64;

  // load Q block (128x64)
#if USE_TDM
  if (wave == 0)
    tdm_load_2d(qs, qg + base + (size_t)q0 * 64, 64, 128, 64, 4);
#else
#pragma unroll
  for (int it = 0; it < 4; ++it) {
    int i = tid + it * 256;
    int r = i >> 3, c = i & 7;
    *(uint4*)(qs + r * 72 + c * 8) =
        *(const uint4*)(qg + base + (size_t)(q0 + r) * 64 + c * 8);
  }
#endif

  float mrow[8], lrow[8];
  v8f o[4];
#pragma unroll
  for (int r = 0; r < 8; ++r) { mrow[r] = -3.0e38f; lrow[r] = 0.f; }
#pragma unroll
  for (int dt = 0; dt < 4; ++dt) o[dt] = zero8();

  const float scale = 0.03125f;              // dim^-0.5 = 1/32
  bf16* pw = ps + wave * 16 * 72;

  for (int kbk = 0; kbk < 32; ++kbk) {
    const int j0 = kbk * 64;
    __syncthreads();
    // K tile 64x64
#if USE_TDM
    if (wave == 0)
      tdm_load_2d(ks, kg + base + (size_t)j0 * 64, 64, 64, 64, 4);
#else
#pragma unroll
    for (int it = 0; it < 2; ++it) {
      int i = tid + it * 256;
      int j = i >> 3, c = i & 7;
      *(uint4*)(ks + j * 72 + c * 8) =
          *(const uint4*)(kg + base + (size_t)(j0 + j) * 64 + c * 8);
    }
#endif
    // V tile, transposed into vt[d][j] (TDM cannot transpose; manual path)
    for (int u = tid; u < 1024; u += 256) {
      int j = u >> 4, d0 = (u & 15) << 2;
      bf16 tmp[4];
      *(uint2*)tmp = *(const uint2*)(vg + base + (size_t)(j0 + j) * 64 + d0);
#pragma unroll
      for (int z = 0; z < 4; ++z) vt[(d0 + z) * 72 + j] = tmp[z];
    }
#if USE_TDM
    if (wave == 0) __builtin_amdgcn_s_wait_tensorcnt(0);  // covers Q + K
#endif
    __syncthreads();

    // S = Q K^T  (16 rows x 64 keys per wave)
    v8f s[4];
#pragma unroll
    for (int jt = 0; jt < 4; ++jt) {
      s[jt] = zero8();
#pragma unroll
      for (int kk = 0; kk < 64; kk += 32) {
        v16bf a = load_frag_a(qs + (wave * 16) * 72, 72, lane, kk);
        v16bf b = load_frag_b(ks + (jt * 16) * 72, 72, lane, kk);
        s[jt] = wmma_bf16(a, b, s[jt]);
      }
#pragma unroll
      for (int r = 0; r < 8; ++r) s[jt][r] *= scale;
    }

    // online softmax (row stats replicated across each 16-lane half)
#pragma unroll
    for (int r = 0; r < 8; ++r) {
      float mx = fmaxf(fmaxf(s[0][r], s[1][r]), fmaxf(s[2][r], s[3][r]));
#pragma unroll
      for (int msk = 8; msk >= 1; msk >>= 1)
        mx = fmaxf(mx, __shfl_xor(mx, msk, 32));
      float mn   = fmaxf(mrow[r], mx);
      float corr = __expf(mrow[r] - mn);
      mrow[r] = mn;
      float rs = 0.f;
#pragma unroll
      for (int jt = 0; jt < 4; ++jt) {
        float p = __expf(s[jt][r] - mn);
        s[jt][r] = p;
        rs += p;
      }
#pragma unroll
      for (int msk = 8; msk >= 1; msk >>= 1) rs += __shfl_xor(rs, msk, 32);
      lrow[r] = lrow[r] * corr + rs;
#pragma unroll
      for (int dt = 0; dt < 4; ++dt) o[dt][r] *= corr;
    }

    // P (C/D layout) -> per-wave LDS (row-major) for A-fragment reload
    {
      int prow = (lane >> 4) << 3;
      int pcol = lane & 15;
#pragma unroll
      for (int jt = 0; jt < 4; ++jt)
#pragma unroll
        for (int r = 0; r < 8; ++r)
          pw[(r + prow) * 72 + jt * 16 + pcol] = (bf16)s[jt][r];
    }
    __syncthreads();

    // O += P * V
#pragma unroll
    for (int dt = 0; dt < 4; ++dt)
#pragma unroll
      for (int kk = 0; kk < 64; kk += 32) {
        v16bf a = load_frag_a(pw, 72, lane, kk);
        v16bf b = load_frag_b(vt + (dt * 16) * 72, 72, lane, kk);
        o[dt] = wmma_bf16(a, b, o[dt]);
      }
  }

  // epilogue: normalize and store to [B, T, H*DH] bf16
  const int b = bh >> 4, h = bh & 15;
  const int prow = (lane >> 4) << 3;
  const int ncol = lane & 15;
#pragma unroll
  for (int dt = 0; dt < 4; ++dt)
#pragma unroll
    for (int r = 0; r < 8; ++r) {
      int grow = q0 + wave * 16 + r + prow;
      int d = dt * 16 + ncol;
      og[(size_t)(b * 2048 + grow) * 1024 + h * 64 + d] =
          (bf16)(o[dt][r] / lrow[r]);
    }
}

// ---------------------------------------------------------------------------
// Output projection: out[m,n] = A[m,:] . Wout[n,:] + bias[n]  (fp32 out)
// ---------------------------------------------------------------------------
__global__ __launch_bounds__(256) void out_gemm_kernel(
    const bf16* __restrict__ A, const bf16* __restrict__ W,
    const float* __restrict__ bias, float* __restrict__ out) {
  __shared__ __align__(16) bf16 as[128 * 40];
  __shared__ __align__(16) bf16 bs[128 * 40];
  v8f acc[2][4];
  const int m0 = blockIdx.x * 128;
  const int n0 = blockIdx.y * 128;
  gemm_mainloop(A, W, 1024, m0, n0, as, bs, acc);

  const int lane = threadIdx.x & 31;
  const int wave = threadIdx.x >> 5;
  const int wm = wave >> 1, wn = wave & 1;
  const int mhi = (lane >> 4) << 3;
#pragma unroll
  for (int mi = 0; mi < 2; ++mi)
#pragma unroll
    for (int nj = 0; nj < 4; ++nj) {
      int col = n0 + wn * 64 + nj * 16 + (lane & 15);
      float bv = bias[col];
#pragma unroll
      for (int r = 0; r < 8; ++r) {
        int row = m0 + wm * 32 + mi * 16 + r + mhi;
        out[(size_t)row * 1024 + col] = acc[mi][nj][r] + bv;
      }
    }
}

// ---------------------------------------------------------------------------
extern "C" void kernel_launch(void* const* d_in, const int* in_sizes, int n_in,
                              void* d_out, int out_size, void* d_ws,
                              size_t ws_size, hipStream_t stream) {
  const float* x     = (const float*)d_in[0];  // [4,2048,1024]
  const float* w_qkv = (const float*)d_in[1];  // [3072,1024]
  const float* w_out = (const float*)d_in[2];  // [1024,1024]
  const float* b_out = (const float*)d_in[3];  // [1024]
  float* out = (float*)d_out;                  // [4,2048,1024]

  char* ws = (char*)d_ws;
  size_t off = 0;
  auto alloc_bf = [&](size_t elems) {
    bf16* p = (bf16*)(ws + off);
    off += elems * sizeof(bf16);
    return p;
  };
  bf16* xb  = alloc_bf(8192ull * 1024);  // x bf16
  bf16* wqb = alloc_bf(3072ull * 1024);  // w_qkv bf16
  bf16* wob = alloc_bf(1024ull * 1024);  // w_out bf16
  bf16* qb  = alloc_bf(8192ull * 1024);  // q [B,H,T,DH]
  bf16* kb  = alloc_bf(8192ull * 1024);  // k
  bf16* vb  = alloc_bf(8192ull * 1024);  // v
  bf16* ab  = alloc_bf(8192ull * 1024);  // attn out [B,T,H*DH]

  cvt_bf16_kernel<<<(8388608 + 255) / 256, 256, 0, stream>>>(x, xb, 8388608);
  cvt_bf16_kernel<<<(3145728 + 255) / 256, 256, 0, stream>>>(w_qkv, wqb, 3145728);
  cvt_bf16_kernel<<<(1048576 + 255) / 256, 256, 0, stream>>>(w_out, wob, 1048576);

  qkv_gemm_kernel<<<dim3(64, 24), 256, 0, stream>>>(xb, wqb, qb, kb, vb);
  attn_kernel<<<dim3(16, 64), 256, 0, stream>>>(qb, kb, vb, ab);
  out_gemm_kernel<<<dim3(64, 8), 256, 0, stream>>>(ab, wob, b_out, out);
}
